// CAM_62852551409742
// MI455X (gfx1250) — matrix operand order repaired and checked by reading, not compiled, not run on hardware
//
#include <hip/hip_runtime.h>
#include <hip/hip_bf16.h>

typedef __attribute__((ext_vector_type(16))) _Float16 v16h;
typedef __attribute__((ext_vector_type(8)))  _Float16 v8h;
typedef __attribute__((ext_vector_type(4)))  _Float16 v4h;
typedef __attribute__((ext_vector_type(8)))  float    v8f;

#define R_   4
#define B_   512
#define T_   150
#define H_   32
#define K_   600      // R*T contraction length
#define KP_  640      // K padded to 20 chunks of 32
#define KROW_ 648     // LDS row stride (halfs) to de-conflict banks
#define NB_  4        // b-values per block (Wc reused 4x from LDS)
#define BLK1 640      // 20 waves: 5 per b-value, each owns 2 M-tiles

// Native CDNA5 TRANS-unit tanh.
#if __has_builtin(__builtin_amdgcn_tanhf)
#define TANHF(x) __builtin_amdgcn_tanhf(x)
#else
#define TANHF(x) tanhf(x)
#endif

#if __has_builtin(__builtin_amdgcn_tanhh)
#define TANH16(x) __builtin_amdgcn_tanhh(x)
#define USE_TANH16 1
#elif __has_builtin(__builtin_amdgcn_tanh_f16)
#define TANH16(x) __builtin_amdgcn_tanh_f16(x)
#define USE_TANH16 1
#else
#define USE_TANH16 0
#endif

// Stage 1: per (r, 4-b-group) workgroup. For each b: GEMM
// A(150x600)@Wc[r]^T(600x32), A[t,k] = tanh(a[r]*f[r,b,t]*feat_n[b,k])
// generated in-register; fused relu(+f*W), dot Wh, +f.
__global__ __launch_bounds__(BLK1) void CAM_stage1(
    const float* __restrict__ feats,  // (R,B,1,T)
    const float* __restrict__ a,      // (R)
    const float* __restrict__ W,      // (R,H)
    const float* __restrict__ Wc,     // (R,H,K)
    const float* __restrict__ Wh,     // (R,H)
    float* __restrict__ ff_out)       // (B, R*T) feats_final workspace
{
  __shared__ __align__(16) _Float16 WcH[H_][KROW_];   // ~41 KB, shared by 4 b
#if USE_TANH16
  __shared__ __align__(16) _Float16 xsh[NB_][KP_];    // feat_n in f16, 5 KB
#else
  __shared__ __align__(16) float    xsf[NB_][KP_];    // feat_n in f32, 10 KB
#endif
  __shared__ float sf[NB_][160];                      // a[r]*f, T padded
  __shared__ float ffv[NB_][160];                     // f[r,b,t]
  __shared__ float Wsh[H_];
  __shared__ float Whsh[H_];

  const int r   = blockIdx.x / (B_ / NB_);
  const int b0  = (blockIdx.x % (B_ / NB_)) * NB_;
  const int tid = threadIdx.x;

  // ---- staging (bases 16B-aligned: 600*4B = 2400B multiples) ----
  {
    const float4* wc4 = (const float4*)(Wc + (size_t)r * H_ * K_);
    for (int idx = tid; idx < H_ * (K_ / 4); idx += BLK1) {     // 4800 f4
      int c = idx / (K_ / 4), i4 = idx % (K_ / 4);
      float4 v = wc4[(size_t)c * (K_ / 4) + i4];
      v4h h; h[0] = (_Float16)v.x; h[1] = (_Float16)v.y;
             h[2] = (_Float16)v.z; h[3] = (_Float16)v.w;
      *(v4h*)&WcH[c][i4 * 4] = h;
    }
    for (int idx = tid; idx < H_ * (KP_ - K_); idx += BLK1) {   // K pad
      int c = idx / (KP_ - K_), k = K_ + idx % (KP_ - K_);
      WcH[c][k] = (_Float16)0.f;
    }
    for (int idx = tid; idx < NB_ * (K_ / 4); idx += BLK1) {    // 600 f4
      int bs = idx / (K_ / 4), i4 = idx % (K_ / 4);
      float4 v = ((const float4*)(feats + (size_t)(b0 + bs) * K_))[i4];
#if USE_TANH16
      v4h h; h[0] = (_Float16)v.x; h[1] = (_Float16)v.y;
             h[2] = (_Float16)v.z; h[3] = (_Float16)v.w;
      *(v4h*)&xsh[bs][i4 * 4] = h;
#else
      *(float4*)&xsf[bs][i4 * 4] = v;
#endif
    }
    if (tid < NB_ * (KP_ - K_)) {                               // x pad
      int bs = tid / (KP_ - K_), k = K_ + tid % (KP_ - K_);
#if USE_TANH16
      xsh[bs][k] = (_Float16)0.f;
#else
      xsf[bs][k] = 0.f;
#endif
    }
    const float ar = a[r];
    if (tid < NB_ * 160) {
      int bs = tid / 160, t = tid % 160;
      float fv = (t < T_) ? feats[(size_t)(r * B_ + b0 + bs) * T_ + t] : 0.f;
      ffv[bs][t] = fv;
      sf[bs][t]  = ar * fv;
    }
    if (tid < H_) { Wsh[tid] = W[r * H_ + tid]; Whsh[tid] = Wh[r * H_ + tid]; }
  }
  __syncthreads();

  const int wv   = tid >> 5;        // 0..19
  const int bs   = wv / 5;          // which b in the group
  const int wsub = wv % 5;          // 2 M-tiles per wave: rows wsub*32..+31
  const int l    = tid & 31;
  const int hh   = l >> 4;          // half-wave (K sub-block per ISA layouts)
  const int lm   = l & 15;
  const int m0   = wsub * 32;

  const float sA = sf[bs][m0 + lm];        // tile A rows m0..m0+15
  const float sB = sf[bs][m0 + 16 + lm];   // tile B rows m0+16..m0+31

  v8f accA0 = {0.f,0.f,0.f,0.f,0.f,0.f,0.f,0.f};
  v8f accA1 = {0.f,0.f,0.f,0.f,0.f,0.f,0.f,0.f};
  v8f accB0 = {0.f,0.f,0.f,0.f,0.f,0.f,0.f,0.f};
  v8f accB1 = {0.f,0.f,0.f,0.f,0.f,0.f,0.f,0.f};

#if USE_TANH16
  const _Float16 sA16 = (_Float16)sA, sB16 = (_Float16)sB;
  v8h svA, svB;
#pragma unroll
  for (int j = 0; j < 8; ++j) { svA[j] = sA16; svB[j] = sB16; }
#endif

#pragma unroll 2
  for (int kc = 0; kc < KP_ / 32; ++kc) {
    const int k0    = kc * 32;
    const int kbase = k0 + hh * 8;   // lanes>=16 own K {8..15, 24..31}
    v16h afA, afB;
#if USE_TANH16
    // Two contiguous 16B runs per lane; v_pk_mul_f16 + v_tanh_f16.
    v8h xlo = *(const v8h*)&xsh[bs][kbase];
    v8h xhi = *(const v8h*)&xsh[bs][kbase + 16];
    v8h pAlo = svA * xlo, pAhi = svA * xhi;
    v8h pBlo = svB * xlo, pBhi = svB * xhi;
#pragma unroll
    for (int j = 0; j < 8; ++j) {
      afA[j]     = TANH16(pAlo[j]);
      afA[8 + j] = TANH16(pAhi[j]);
      afB[j]     = TANH16(pBlo[j]);
      afB[8 + j] = TANH16(pBhi[j]);
    }
#else
#pragma unroll
    for (int j = 0; j < 16; ++j) {
      int k = kbase + (j < 8 ? j : j + 8);      // {0..7,16..23} pattern
      float x = xsf[bs][k];
      afA[j] = (_Float16)TANHF(sA * x);
      afB[j] = (_Float16)TANHF(sB * x);
    }
#endif
    // B fragment: lane = column n, K {8h..8h+7, 16+8h..16+8h+7} from LDS.
    const int kB = k0 + hh * 8;
    v8h b0lo = *(const v8h*)&WcH[lm][kB], b0hi = *(const v8h*)&WcH[lm][kB + 16];
    v8h b1lo = *(const v8h*)&WcH[lm + 16][kB], b1hi = *(const v8h*)&WcH[lm + 16][kB + 16];
    v16h bf0 = __builtin_shufflevector(b0lo, b0hi, 0,1,2,3,4,5,6,7,8,9,10,11,12,13,14,15);        // N = 0..15
    v16h bf1 = __builtin_shufflevector(b1lo, b1hi, 0,1,2,3,4,5,6,7,8,9,10,11,12,13,14,15);   // N = 16..31

    accA0 = __builtin_amdgcn_wmma_f32_16x16x32_f16(
        false, afA, false, bf0, (short)0, accA0, false, false);
    accA1 = __builtin_amdgcn_wmma_f32_16x16x32_f16(
        false, afA, false, bf1, (short)0, accA1, false, false);
    accB0 = __builtin_amdgcn_wmma_f32_16x16x32_f16(
        false, afB, false, bf0, (short)0, accB0, false, false);
    accB1 = __builtin_amdgcn_wmma_f32_16x16x32_f16(
        false, afB, false, bf1, (short)0, accB1, false, false);
  }

  // Epilogue: Hm = relu(G + f*W); attf = Hm . Wh + f.
  const float w0  = Wsh[lm],  w1  = Wsh[lm + 16];
  const float wh0 = Whsh[lm], wh1 = Whsh[lm + 16];
  float* outp = ff_out + (size_t)(b0 + bs) * (R_ * T_) + r * T_;

#pragma unroll
  for (int i = 0; i < 8; ++i) {
    const int mrow = i + hh * 8;   // C layout: lanes16-31 hold M=i+8
    {   // tile A (rows m0 .. m0+15)
      const int t = m0 + mrow;
      const float fval = ffv[bs][t];
      float g0 = accA0[i] + fval * w0; g0 = g0 > 0.f ? g0 : 0.f;
      float g1 = accA1[i] + fval * w1; g1 = g1 > 0.f ? g1 : 0.f;
      float v = g0 * wh0 + g1 * wh1;
      v += __shfl_xor(v, 8);
      v += __shfl_xor(v, 4);
      v += __shfl_xor(v, 2);
      v += __shfl_xor(v, 1);
      if (lm == 0 && t < T_) outp[t] = v + fval;
    }
    {   // tile B (rows m0+16 .. m0+31)
      const int t = m0 + 16 + mrow;
      const float fval = ffv[bs][t];
      float g0 = accB0[i] + fval * w0; g0 = g0 > 0.f ? g0 : 0.f;
      float g1 = accB1[i] + fval * w1; g1 = g1 > 0.f ? g1 : 0.f;
      float v = g0 * wh0 + g1 * wh1;
      v += __shfl_xor(v, 8);
      v += __shfl_xor(v, 4);
      v += __shfl_xor(v, 2);
      v += __shfl_xor(v, 1);
      if (lm == 0 && t < T_) outp[t] = v + fval;
    }
  }
}

// Stage 2: out = (feats_final @ W1^T + b1) @ W2^T + b2.
// 32 blocks x 16 batch rows: W1 read once per block, ff tile in LDS.
#define BB2 16
__global__ __launch_bounds__(128) void CAM_stage2(
    const float* __restrict__ ff,   // (B, 600)
    const float* __restrict__ W1,   // (128, 600)
    const float* __restrict__ b1,   // (128)
    const float* __restrict__ W2,   // (7, 128)
    const float* __restrict__ b2,   // (7)
    float* __restrict__ out)        // (B, 1, 7)
{
  __shared__ __align__(16) float fft[BB2][K_];   // 38.4 KB
  __shared__ float hsh[BB2][128];                // 8 KB
  const int b0  = blockIdx.x * BB2;
  const int tid = threadIdx.x;

  const float4* f4 = (const float4*)(ff + (size_t)b0 * K_);
  float4* s4 = (float4*)&fft[0][0];
  for (int i = tid; i < BB2 * (K_ / 4); i += 128) s4[i] = f4[i];
  __syncthreads();

  const int c = tid;
  float acc[BB2];
  const float bc = b1[c];
#pragma unroll
  for (int j = 0; j < BB2; ++j) acc[j] = bc;

  const float* w = W1 + (size_t)c * K_;
  for (int k = 0; k < K_; ++k) {
    const float wv = w[k];
#pragma unroll
    for (int j = 0; j < BB2; ++j) acc[j] += fft[j][k] * wv;
  }
#pragma unroll
  for (int j = 0; j < BB2; ++j) hsh[j][c] = acc[j];
  __syncthreads();

  if (tid < BB2 * 7) {
    const int j = tid / 7, o = tid % 7;
    float v = b2[o];
    const float* w2 = W2 + o * 128;
#pragma unroll 16
    for (int q = 0; q < 128; ++q) v += hsh[j][q] * w2[q];
    out[(size_t)(b0 + j) * 7 + o] = v;
  }
}

extern "C" void kernel_launch(void* const* d_in, const int* in_sizes, int n_in,
                              void* d_out, int out_size, void* d_ws, size_t ws_size,
                              hipStream_t stream) {
  const float* feats = (const float*)d_in[0];
  const float* a     = (const float*)d_in[1];
  const float* W     = (const float*)d_in[2];
  const float* Wc    = (const float*)d_in[3];
  const float* Wh    = (const float*)d_in[4];
  const float* W1    = (const float*)d_in[5];
  const float* b1    = (const float*)d_in[6];
  const float* W2    = (const float*)d_in[7];
  const float* b2    = (const float*)d_in[8];
  float* out = (float*)d_out;
  float* ffw = (float*)d_ws;   // B*600 floats = 1.2 MB scratch

  CAM_stage1<<<dim3(R_ * (B_ / NB_)), dim3(BLK1), 0, stream>>>(
      feats, a, W, Wc, Wh, ffw);
  CAM_stage2<<<dim3(B_ / BB2), dim3(128), 0, stream>>>(ffw, W1, b1, W2, b2, out);
}
